// DirectVoxGO_17239998726730
// MI455X (gfx1250) — compile-verified, matrix operand order checked by reading, not run on hardware
//
#include <hip/hip_runtime.h>
#include <hip/hip_bf16.h>
#include <math.h>

typedef __attribute__((ext_vector_type(16))) _Float16 v16h;
typedef __attribute__((ext_vector_type(8)))  float    v8f;

#define GRES   160
#define GXY    (160*160)        // x-stride
#define GVOL   (160*160*160)    // channel stride
#define NS     256
#define ACT_SHIFT_F (-13.8155096f)   // log(1/(1-1e-6) - 1)

// d_ws layout:
//   [0 .. P*16)            float rgba[P][4]  = {r,g,b,alpha}
//   [P*16 .. )             f16 transposed weights (26624 halves):
//        w0T[128][64]   (K padded 39->64)      8192 halves
//        w1T[128][128]                         16384 halves
//        w2T[16][128]   (N padded 3->16)       2048 halves

__global__ void prep_weights(const float* __restrict__ w0,
                             const float* __restrict__ w1,
                             const float* __restrict__ w2,
                             _Float16* __restrict__ wts) {
  int i = blockIdx.x * 256 + threadIdx.x;
  if (i < 8192) {                       // w0T: [n][k], k padded to 64
    int n = i >> 6, k = i & 63;
    wts[i] = (_Float16)((k < 39) ? w0[k * 128 + n] : 0.0f);
  } else if (i < 24576) {               // w1T: [n][k]
    int j = i - 8192; int n = j >> 7, k = j & 127;
    wts[i] = (_Float16)w1[k * 128 + n];
  } else if (i < 26624) {               // w2T: [n][k], n padded to 16
    int j = i - 24576; int n = j >> 7, k = j & 127;
    wts[i] = (_Float16)((n < 3) ? w2[k * 3 + n] : 0.0f);
  }
}

__global__ __launch_bounds__(256)
void fused_point_mlp(const float* __restrict__ rays_o,
                     const float* __restrict__ rays_d,
                     const float* __restrict__ dens,
                     const float* __restrict__ feat,
                     const float* __restrict__ b0,
                     const float* __restrict__ b1,
                     const float* __restrict__ b2,
                     const _Float16* __restrict__ wts,
                     float* __restrict__ rgba) {
  // 100 KB LDS total (CDNA5 WGP has 320 KB; ~3 workgroups resident)
  __shared__ __align__(32) _Float16 sW[26624];       // staged weights (52 KB)
  __shared__ __align__(32) _Float16 sX[128 * 64];    // activation tile (16 KB)
  __shared__ __align__(32) _Float16 sH[128 * 128];   // hidden tile    (32 KB)

  const int tid = threadIdx.x;
  const int p0  = blockIdx.x * 128;

  // ---------------- Stage weights to LDS (all 256 threads) ------------------
  {
    const uint4* src = (const uint4*)wts;
    uint4*       dst = (uint4*)sW;
#pragma unroll
    for (int i = 0; i < 13; ++i) {          // 13 * 256 = 3328 uint4 = 53248 B
      dst[i * 256 + tid] = src[i * 256 + tid];
    }
  }

  // ---------------- Phase A: trilerp + embedding (2 threads / point) -------
  {
    const int half = tid >> 7;          // which half of the point's work
    const int pl   = tid & 127;         // point within block
    const int p    = p0 + pl;
    const int r = p >> 8, s = p & 255;
    float ox = rays_o[3*r], oy = rays_o[3*r+1], oz = rays_o[3*r+2];
    float dx = rays_d[3*r], dy = rays_d[3*r+1], dz = rays_d[3*r+2];
    float inv = rsqrtf(dx*dx + dy*dy + dz*dz);
    dx *= inv; dy *= inv; dz *= inv;
    float t = 0.05f + (float)s * (1.95f / 255.0f);
    float px = ox + dx*t, py = oy + dy*t, pz = oz + dz*t;
    bool inbox = (px >= -1.f && px <= 1.f && py >= -1.f && py <= 1.f &&
                  pz >= -1.f && pz <= 1.f);

    float ix = fminf(fmaxf((px + 1.f) * 0.5f * 159.f, 0.f), 159.f);
    float iy = fminf(fmaxf((py + 1.f) * 0.5f * 159.f, 0.f), 159.f);
    float iz = fminf(fmaxf((pz + 1.f) * 0.5f * 159.f, 0.f), 159.f);
    int x0 = (int)ix; if (x0 > 158) x0 = 158;
    int y0 = (int)iy; if (y0 > 158) y0 = 158;
    int z0 = (int)iz; if (z0 > 158) z0 = 158;
    float fx = ix - (float)x0, fy = iy - (float)y0, fz = iz - (float)z0;
    float gx = 1.f - fx, gy = 1.f - fy, gz = 1.f - fz;
    float w000 = gx*gy*gz, w100 = fx*gy*gz, w010 = gx*fy*gz, w001 = gx*gy*fz;
    float w110 = fx*fy*gz, w101 = fx*gy*fz, w011 = gx*fy*fz, w111 = fx*fy*fz;
    int idx = (x0 * GRES + y0) * GRES + z0;
    _Float16* xr = &sX[pl * 64];

    if (half == 0) {
      const float* g = dens + idx;
      float d = g[0]       * w000 + g[GXY]        * w100 +
                g[GRES]    * w010 + g[1]          * w001 +
                g[GXY+GRES]* w110 + g[GXY+1]      * w101 +
                g[GRES+1]  * w011 + g[GXY+GRES+1] * w111;
      float alpha = inbox ? (1.f - rsqrtf(1.f + expf(d + ACT_SHIFT_F))) : 0.f;
      rgba[(size_t)p * 4 + 3] = alpha;
#pragma unroll
      for (int c = 0; c < 6; ++c) {
        const float* gc = feat + c * GVOL + idx;
        float v = gc[0]       * w000 + gc[GXY]        * w100 +
                  gc[GRES]    * w010 + gc[1]          * w001 +
                  gc[GXY+GRES]* w110 + gc[GXY+1]      * w101 +
                  gc[GRES+1]  * w011 + gc[GXY+GRES+1] * w111;
        xr[c] = (_Float16)v;
      }
    } else {
#pragma unroll
      for (int c = 6; c < 12; ++c) {
        const float* gc = feat + c * GVOL + idx;
        float v = gc[0]       * w000 + gc[GXY]        * w100 +
                  gc[GRES]    * w010 + gc[1]          * w001 +
                  gc[GXY+GRES]* w110 + gc[GXY+1]      * w101 +
                  gc[GRES+1]  * w011 + gc[GXY+GRES+1] * w111;
        xr[c] = (_Float16)v;
      }
      xr[12] = (_Float16)dx; xr[13] = (_Float16)dy; xr[14] = (_Float16)dz;
      float vd3[3] = {dx, dy, dz};
#pragma unroll
      for (int i = 0; i < 3; ++i) {
        float f = 1.f;
#pragma unroll
        for (int j = 0; j < 4; ++j) {
          float vf = vd3[i] * f, sv, cv;
          __sincosf(vf, &sv, &cv);
          xr[15 + i*4 + j] = (_Float16)sv;
          xr[27 + i*4 + j] = (_Float16)cv;
          f *= 2.f;
        }
      }
#pragma unroll
      for (int k = 39; k < 64; ++k) xr[k] = (_Float16)0.0f;  // K padding
    }
  }
  __syncthreads();

  // ---------------- Phase B: WMMA MLP, one 16-row M-tile per wave -----------
  const int wave  = tid >> 5;
  const int lane  = tid & 31;
  const int nrow  = lane & 15;     // M row for A frag; N col for B/C frags
  const int khalf = lane >> 4;     // K half for A/B frags; M half for C/D

  const _Float16* w0T = sW;
  const _Float16* w1T = sW + 8192;
  const _Float16* w2T = sW + 24576;
  const _Float16* aX  = &sX[(wave * 16 + nrow) * 64];
  _Float16*       hB  = &sH[(wave * 16) * 128];
  const _Float16* aH  = &sH[(wave * 16 + nrow) * 128];

  v8f acc[8];

  // Layer 0: [16x64] @ [64x128]  (16 WMMAs)
#pragma unroll
  for (int nt = 0; nt < 8; ++nt) {
    float bv = b0[nt * 16 + nrow];
    v8f c;
#pragma unroll
    for (int q = 0; q < 8; ++q) c[q] = bv;
#pragma unroll
    for (int kt = 0; kt < 2; ++kt) {
      v16h a = *(const v16h*)(aX + kt * 32 + khalf * 16);
      v16h b = *(const v16h*)(w0T + (nt * 16 + nrow) * 64 + kt * 32 + khalf * 16);
      c = __builtin_amdgcn_wmma_f32_16x16x32_f16(false, a, false, b,
                                                 (short)0, c, false, false);
    }
    acc[nt] = c;
  }
#pragma unroll
  for (int nt = 0; nt < 8; ++nt)
#pragma unroll
    for (int q = 0; q < 8; ++q)
      hB[(khalf * 8 + q) * 128 + nt * 16 + nrow] =
          (_Float16)fmaxf(acc[nt][q], 0.f);
  __syncthreads();

  // Layer 1: [16x128] @ [128x128]  (32 WMMAs)
#pragma unroll
  for (int nt = 0; nt < 8; ++nt) {
    float bv = b1[nt * 16 + nrow];
    v8f c;
#pragma unroll
    for (int q = 0; q < 8; ++q) c[q] = bv;
#pragma unroll
    for (int kt = 0; kt < 4; ++kt) {
      v16h a = *(const v16h*)(aH + kt * 32 + khalf * 16);
      v16h b = *(const v16h*)(w1T + (nt * 16 + nrow) * 128 + kt * 32 + khalf * 16);
      c = __builtin_amdgcn_wmma_f32_16x16x32_f16(false, a, false, b,
                                                 (short)0, c, false, false);
    }
    acc[nt] = c;
  }
#pragma unroll
  for (int nt = 0; nt < 8; ++nt)
#pragma unroll
    for (int q = 0; q < 8; ++q)
      hB[(khalf * 8 + q) * 128 + nt * 16 + nrow] =
          (_Float16)fmaxf(acc[nt][q], 0.f);
  __syncthreads();

  // Layer 2: [16x128] @ [128x16(pad of 3)]  (4 WMMAs)
  {
    float bv = (nrow < 3) ? b2[nrow] : 0.f;
    v8f c;
#pragma unroll
    for (int q = 0; q < 8; ++q) c[q] = bv;
#pragma unroll
    for (int kt = 0; kt < 4; ++kt) {
      v16h a = *(const v16h*)(aH + kt * 32 + khalf * 16);
      v16h b = *(const v16h*)(w2T + nrow * 128 + kt * 32 + khalf * 16);
      c = __builtin_amdgcn_wmma_f32_16x16x32_f16(false, a, false, b,
                                                 (short)0, c, false, false);
    }
    if (nrow < 3) {
#pragma unroll
      for (int q = 0; q < 8; ++q) {
        int m = khalf * 8 + q;
        int p = p0 + wave * 16 + m;
        rgba[(size_t)p * 4 + nrow] = 1.f / (1.f + expf(-c[q]));
      }
    }
  }
}

__global__ __launch_bounds__(256)
void composite(const float* __restrict__ rgba, float* __restrict__ out) {
  const int r = blockIdx.x, s = threadIdx.x;
  __shared__ float cp[256];
  __shared__ float red[256];
  const float* q = rgba + ((size_t)r * 256 + s) * 4;
  float c0 = q[0], c1 = q[1], c2 = q[2], a = q[3];
  cp[s] = 1.f - a;
  __syncthreads();
#pragma unroll
  for (int off = 1; off < 256; off <<= 1) {   // inclusive product scan
    float u = (s >= off) ? cp[s - off] : 1.f;
    float v = cp[s];
    __syncthreads();
    cp[s] = u * v;
    __syncthreads();
  }
  float Texcl = (s == 0) ? 1.f : cp[s - 1];
  float wgt   = a * Texcl;
  float last  = cp[255];                       // alphainv_last
  float rgb3[3] = {c0, c1, c2};
#pragma unroll
  for (int ch = 0; ch < 3; ++ch) {
    red[s] = wgt * rgb3[ch];
    __syncthreads();
    for (int st = 128; st > 0; st >>= 1) {
      if (s < st) red[s] += red[s + st];
      __syncthreads();
    }
    if (s == 0) out[r * 3 + ch] = red[0] + last;
    __syncthreads();
  }
}

extern "C" void kernel_launch(void* const* d_in, const int* in_sizes, int n_in,
                              void* d_out, int out_size, void* d_ws, size_t ws_size,
                              hipStream_t stream) {
  const float* rays_o  = (const float*)d_in[0];
  const float* rays_d  = (const float*)d_in[1];
  const float* density = (const float*)d_in[2];
  const float* k0      = (const float*)d_in[3];
  const float* w0      = (const float*)d_in[4];
  const float* b0      = (const float*)d_in[5];
  const float* w1      = (const float*)d_in[6];
  const float* b1      = (const float*)d_in[7];
  const float* w2      = (const float*)d_in[8];
  const float* b2      = (const float*)d_in[9];

  const int N = in_sizes[0] / 3;       // rays
  const int P = N * NS;                // points

  float*    rgba = (float*)d_ws;
  _Float16* wts  = (_Float16*)((char*)d_ws + (size_t)P * 4 * sizeof(float));

  prep_weights<<<(26624 + 255) / 256, 256, 0, stream>>>(w0, w1, w2, wts);
  fused_point_mlp<<<P / 128, 256, 0, stream>>>(rays_o, rays_d, density, k0,
                                               b0, b1, b2, wts, rgba);
  composite<<<N, 256, 0, stream>>>(rgba, (float*)d_out);
}
